// SPM_backbone_12446815224487
// MI455X (gfx1250) — matrix operand order repaired and checked by reading, not compiled, not run on hardware
//
#include <hip/hip_runtime.h>
#include <math.h>

// ---------------- types ----------------
typedef __bf16 bf16;
typedef __attribute__((ext_vector_type(16))) __bf16 v16bf;
typedef __attribute__((ext_vector_type(8)))  __bf16 v8bf;
typedef __attribute__((ext_vector_type(8)))  float  v8f;

// ---------------- problem constants ----------------
#define BB    64
#define CC    862
#define FBN   257
#define DM    256
#define DFF   512
#define KP    288              // padded K per complex half (mult. of 32)
#define KEMB  (2*KP)           // 576
#define MEMB  (BB*CC)          // 55168  (divisible by 128)
#define MTOK  (2*BB*CC)        // 110336 (divisible by 128)
#define NYE   ((size_t)MTOK*DM)// 28,246,016 elements in y
#define EPSB  1e-5f

// ---------------- WMMA fragment loads (per ISA 7.12.2 layouts) ----------------
// A (16x32 bf16): lane L holds row M=L&15; K = {(L>>4)*8 .. +7} and {+16 .. +23}
__device__ __forceinline__ v16bf frag_a(const bf16* __restrict__ row, int kb, int lane) {
  const int off = kb + ((lane >> 4) << 3);
  union { v16bf v; v8bf h[2]; } u;
  u.h[0] = *reinterpret_cast<const v8bf*>(row + off);
  u.h[1] = *reinterpret_cast<const v8bf*>(row + off + 16);
  return u.v;
}
// B (32x16 bf16): lane L holds col N=L&15; K = {(L>>4)*8 .. +7} and {+16 .. +23}
__device__ __forceinline__ v16bf frag_b(const bf16* __restrict__ row, int kb, int lane) {
  const int off = kb + ((lane >> 4) << 3);
  union { v16bf v; v8bf h[2]; } u;
  u.h[0] = *reinterpret_cast<const v8bf*>(row + off);
  u.h[1] = *reinterpret_cast<const v8bf*>(row + off + 16);
  return u.v;
}

// Wave computes a 32(M) x 64(N) tile: 8 wmma accumulators.
// Explicit ping-pong double buffering with K-step 64 (K % 64 == 0 for all our
// GEMMs): load set1(k+32) -> WMMA(set0) -> load set0(k+64) -> WMMA(set1).
// The buffers alternate roles structurally, so no inter-iteration register
// copies are needed, while next-step loads stay in flight during the math.
// A row-major [M][K], B row-major [N][K] (i.e. C = A * B^T).
__device__ __forceinline__ void wmma_tile(const bf16* __restrict__ A,
                                          const bf16* __restrict__ Bw,
                                          int lda, int ldb, int K,
                                          int m0, int n0, int lane, v8f acc[2][4]) {
  const bf16* arow0 = A + (size_t)(m0 + (lane & 15)) * lda;
  const bf16* arow1 = arow0 + (size_t)16 * lda;
  const bf16* br0 = Bw + (size_t)(n0 + (lane & 15)) * ldb;
  const bf16* br1 = br0 + (size_t)16 * ldb;
  const bf16* br2 = br0 + (size_t)32 * ldb;
  const bf16* br3 = br0 + (size_t)48 * ldb;

  v16bf a0[2], b0[4], a1[2], b1[4];
  a0[0] = frag_a(arow0, 0, lane);  a0[1] = frag_a(arow1, 0, lane);
  b0[0] = frag_b(br0, 0, lane);    b0[1] = frag_b(br1, 0, lane);
  b0[2] = frag_b(br2, 0, lane);    b0[3] = frag_b(br3, 0, lane);

  int kb = 0;
  for (; kb + 64 < K; kb += 64) {
    const int k1 = kb + 32, k2 = kb + 64;
    a1[0] = frag_a(arow0, k1, lane);  a1[1] = frag_a(arow1, k1, lane);
    b1[0] = frag_b(br0, k1, lane);    b1[1] = frag_b(br1, k1, lane);
    b1[2] = frag_b(br2, k1, lane);    b1[3] = frag_b(br3, k1, lane);
#pragma unroll
    for (int u = 0; u < 2; ++u)
#pragma unroll
      for (int t = 0; t < 4; ++t)
        acc[u][t] = __builtin_amdgcn_wmma_f32_16x16x32_bf16(
            false, a0[u], false, b0[t], (short)0, acc[u][t], false, false);
    a0[0] = frag_a(arow0, k2, lane);  a0[1] = frag_a(arow1, k2, lane);
    b0[0] = frag_b(br0, k2, lane);    b0[1] = frag_b(br1, k2, lane);
    b0[2] = frag_b(br2, k2, lane);    b0[3] = frag_b(br3, k2, lane);
#pragma unroll
    for (int u = 0; u < 2; ++u)
#pragma unroll
      for (int t = 0; t < 4; ++t)
        acc[u][t] = __builtin_amdgcn_wmma_f32_16x16x32_bf16(
            false, a1[u], false, b1[t], (short)0, acc[u][t], false, false);
  }
  // tail: last 64 of K (kb == K - 64 here)
  {
    const int k1 = kb + 32;
    a1[0] = frag_a(arow0, k1, lane);  a1[1] = frag_a(arow1, k1, lane);
    b1[0] = frag_b(br0, k1, lane);    b1[1] = frag_b(br1, k1, lane);
    b1[2] = frag_b(br2, k1, lane);    b1[3] = frag_b(br3, k1, lane);
#pragma unroll
    for (int u = 0; u < 2; ++u)
#pragma unroll
      for (int t = 0; t < 4; ++t)
        acc[u][t] = __builtin_amdgcn_wmma_f32_16x16x32_bf16(
            false, a0[u], false, b0[t], (short)0, acc[u][t], false, false);
#pragma unroll
    for (int u = 0; u < 2; ++u)
#pragma unroll
      for (int t = 0; t < 4; ++t)
        acc[u][t] = __builtin_amdgcn_wmma_f32_16x16x32_bf16(
            false, a1[u], false, b1[t], (short)0, acc[u][t], false, false);
  }
}

// ---------------- GEMM kernels (256 threads = 8 waves; block tile 128M x 128N) --
// Embedding: A=[x_re|x_im] (M=55168,K=576), B=[Wre|-Wim ; Wim|Wre] (N=512,K=576).
// n<256 -> y_re -> y[b], n>=256 -> y_im -> y[b+64].
__global__ __launch_bounds__(256) void gemm_emb_kernel(
    const bf16* __restrict__ A, const bf16* __restrict__ Bw,
    const float* __restrict__ bre, const float* __restrict__ bim,
    float* __restrict__ y) {
  const int lane = threadIdx.x & 31, wave = threadIdx.x >> 5;
  const int m0 = blockIdx.x * 128 + (wave & 3) * 32;
  const int n0 = blockIdx.y * 128 + (wave >> 2) * 64;
  v8f acc[2][4] = {};
  wmma_tile(A, Bw, KEMB, KEMB, KEMB, m0, n0, lane, acc);
#pragma unroll
  for (int u = 0; u < 2; ++u) {
    const int rbase = m0 + u * 16 + ((lane >> 4) << 3);
#pragma unroll
    for (int t = 0; t < 4; ++t) {
      const int n = n0 + t * 16 + (lane & 15);
      const int half = (n >= DM) ? 1 : 0;
      const int nn = n - half * DM;
      const float bias = half ? bim[nn] : bre[nn];
#pragma unroll
      for (int r = 0; r < 8; ++r) {
        const int m = rbase + r;
        const int b = m / CC, c = m % CC;
        y[(((size_t)(b + half * BB)) * CC + c) * DM + nn] = acc[u][t][r] + bias;
      }
    }
  }
}

// FF1: 256 -> 512, + bias, exact GELU, bf16 store
__global__ __launch_bounds__(256) void gemm_ff1_kernel(
    const bf16* __restrict__ A, const bf16* __restrict__ Bw,
    const float* __restrict__ bias1, bf16* __restrict__ ff) {
  const int lane = threadIdx.x & 31, wave = threadIdx.x >> 5;
  const int m0 = blockIdx.x * 128 + (wave & 3) * 32;
  const int n0 = blockIdx.y * 128 + (wave >> 2) * 64;
  v8f acc[2][4] = {};
  wmma_tile(A, Bw, DM, DM, DM, m0, n0, lane, acc);
#pragma unroll
  for (int u = 0; u < 2; ++u) {
    const int rbase = m0 + u * 16 + ((lane >> 4) << 3);
#pragma unroll
    for (int t = 0; t < 4; ++t) {
      const int n = n0 + t * 16 + (lane & 15);
      const float bs = bias1[n];
#pragma unroll
      for (int r = 0; r < 8; ++r) {
        float v = acc[u][t][r] + bs;
        v = 0.5f * v * (1.0f + erff(v * 0.70710678118654752440f));
        ff[(size_t)(rbase + r) * DFF + n] = (bf16)v;
      }
    }
  }
}

// FF2: 512 -> 256, + bias + residual (normalized src), fp32 store
__global__ __launch_bounds__(256) void gemm_ff2_kernel(
    const bf16* __restrict__ A, const bf16* __restrict__ Bw,
    const float* __restrict__ bias2, const float* __restrict__ resid,
    float* __restrict__ out) {
  const int lane = threadIdx.x & 31, wave = threadIdx.x >> 5;
  const int m0 = blockIdx.x * 128 + (wave & 3) * 32;
  const int n0 = blockIdx.y * 128 + (wave >> 2) * 64;
  v8f acc[2][4] = {};
  wmma_tile(A, Bw, DFF, DFF, DFF, m0, n0, lane, acc);
#pragma unroll
  for (int u = 0; u < 2; ++u) {
    const int rbase = m0 + u * 16 + ((lane >> 4) << 3);
#pragma unroll
    for (int t = 0; t < 4; ++t) {
      const int n = n0 + t * 16 + (lane & 15);
      const float bs = bias2[n];
#pragma unroll
      for (int r = 0; r < 8; ++r) {
        const size_t idx = (size_t)(rbase + r) * DM + n;
        out[idx] = acc[u][t][r] + bs + resid[idx];
      }
    }
  }
}

// ---------------- elementwise kernels ----------------
// "attention" (pure elementwise with flip + transpose-reshape gather) + residual.
// src[bb,c',d'] = y[bb,c',d'] + o(bb, f/CC, f%CC), f = c'*256+d'
__global__ __launch_bounds__(256) void attn_kernel(const float* __restrict__ y,
                                                   float* __restrict__ src) {
  const size_t i = (size_t)blockIdx.x * 256 + threadIdx.x;
  const int d = (int)(i & (DM - 1));
  const int rest = (int)(i >> 8);
  const int c = rest % CC;
  const int bb = rest / CC;
  const int f = c * DM + d;          // flat index in (C,D) plane
  const int dg = f / CC;             // gathered d
  const int cg = f % CC;             // gathered c
  const int b = bb & (BB - 1);
  const size_t half = (size_t)BB * CC * DM;
  const size_t pq = ((size_t)b * CC + cg) * DM + dg;
  const size_t pk = ((size_t)b * CC + (CC - 1 - cg)) * DM + (DM - 1 - dg);
  const float qr = y[pq], qi = y[pq + half];
  const float kr = y[pk], ki = y[pk + half];
  const float asr = 0.25f * (qr * kr - qi * ki);   // scale = (256/16)^-0.5
  const float asi = 0.25f * (qr * ki - qi * kr);
  // v_re = v_im = y[:64] at (b,cg,dg) == qr
  const float o = (bb < BB) ? (asr - asi) * qr : (asr + asi) * qr;
  src[i] = y[i] + o;
}

// Deterministic batchnorm reduction: per-block partial sums, ordered finalize.
__global__ __launch_bounds__(256) void bn_reduce_kernel(const float* __restrict__ x,
                                                        float* __restrict__ part) {
  const int d = threadIdx.x;
  float s = 0.f, s2 = 0.f;
  for (int r = blockIdx.x; r < MTOK; r += gridDim.x) {
    const float v = x[(size_t)r * DM + d];
    s += v; s2 += v * v;
  }
  part[(size_t)blockIdx.x * (2 * DM) + d]      = s;
  part[(size_t)blockIdx.x * (2 * DM) + DM + d] = s2;
}

__global__ __launch_bounds__(256) void bn_finalize_kernel(const float* __restrict__ part,
                                                          float* __restrict__ acc,
                                                          int nblk) {
  const int d = threadIdx.x;
  float s = 0.f, s2 = 0.f;
  for (int b = 0; b < nblk; ++b) {
    s  += part[(size_t)b * (2 * DM) + d];
    s2 += part[(size_t)b * (2 * DM) + DM + d];
  }
  acc[d] = s; acc[DM + d] = s2;
}

// normalize in place; optionally emit bf16 copy (GEMM input)
__global__ __launch_bounds__(256) void bn_apply_kernel(float* __restrict__ x,
    const float* __restrict__ acc, const float* __restrict__ g,
    const float* __restrict__ be, bf16* __restrict__ xbf) {
  const size_t i = (size_t)blockIdx.x * 256 + threadIdx.x;
  const int d = (int)(i & (DM - 1));
  const float inv = 1.0f / (float)MTOK;
  const float m = acc[d] * inv;
  const float var = acc[DM + d] * inv - m * m;
  const float sc = rsqrtf(var + EPSB) * g[d];
  const float v = (x[i] - m) * sc + be[d];
  x[i] = v;
  if (xbf) xbf[i] = (bf16)v;
}

// ---------------- prep / output kernels ----------------
__global__ __launch_bounds__(256) void prep_xcat_kernel(const float* __restrict__ xr,
    const float* __restrict__ xi, bf16* __restrict__ xcat) {
  const size_t idx = (size_t)blockIdx.x * 256 + threadIdx.x; // MEMB*KEMB
  const int k = (int)(idx % KEMB);
  const size_t m = idx / KEMB;
  float v = 0.f;
  if (k < KP) { if (k < FBN) v = xr[m * FBN + k]; }
  else { const int kk = k - KP; if (kk < FBN) v = xi[m * FBN + kk]; }
  xcat[idx] = (bf16)v;
}

__global__ __launch_bounds__(256) void prep_wemb_kernel(const float* __restrict__ wre,
    const float* __restrict__ wim, bf16* __restrict__ wemb) {
  const int idx = blockIdx.x * 256 + threadIdx.x; // 512*KEMB
  const int k = idx % KEMB;
  const int n = idx / KEMB;
  float v = 0.f;
  if (n < DM) {  // y_re row: [Wre | -Wim]
    if (k < KP) { if (k < FBN) v = wre[n * FBN + k]; }
    else { const int kk = k - KP; if (kk < FBN) v = -wim[n * FBN + kk]; }
  } else {       // y_im row: [Wim |  Wre]
    const int nn = n - DM;
    if (k < KP) { if (k < FBN) v = wim[nn * FBN + k]; }
    else { const int kk = k - KP; if (kk < FBN) v = wre[nn * FBN + kk]; }
  }
  wemb[idx] = (bf16)v;
}

__global__ __launch_bounds__(256) void cvt_bf16_kernel(const float* __restrict__ in,
                                                       bf16* __restrict__ out, int n) {
  const int i = blockIdx.x * 256 + threadIdx.x;
  if (i < n) out[i] = (bf16)in[i];
}

// out[b,c,d,0]=y[b,c,d]; out[b,c,d,1]=y[b+64,c,d]
__global__ __launch_bounds__(256) void write_out_kernel(const float* __restrict__ y,
                                                        float* __restrict__ out) {
  const size_t i = (size_t)blockIdx.x * 256 + threadIdx.x; // BB*CC*DM
  out[i * 2]     = y[i];
  out[i * 2 + 1] = y[i + (size_t)BB * CC * DM];
}

// ---------------- host ----------------
extern "C" void kernel_launch(void* const* d_in, const int* in_sizes, int n_in,
                              void* d_out, int out_size, void* d_ws, size_t ws_size,
                              hipStream_t stream) {
  (void)in_sizes; (void)n_in; (void)out_size; (void)ws_size;
  const float* x_re = (const float*)d_in[0];
  const float* x_im = (const float*)d_in[1];
  const float* eWre = (const float*)d_in[2];
  const float* eWim = (const float*)d_in[3];
  const float* ebre = (const float*)d_in[4];
  const float* ebim = (const float*)d_in[5];
  const float* W1   = (const float*)d_in[6];
  const float* b1   = (const float*)d_in[7];
  const float* W2   = (const float*)d_in[8];
  const float* b2   = (const float*)d_in[9];
  const float* g1   = (const float*)d_in[10];
  const float* be1  = (const float*)d_in[11];
  const float* g2   = (const float*)d_in[12];
  const float* be2  = (const float*)d_in[13];

  char* ws = (char*)d_ws;
  size_t off = 0;
  auto alloc = [&](size_t bytes) -> void* {
    void* p = ws + off;
    off = (off + bytes + 255) & ~(size_t)255;
    return p;
  };
  float* ybuf   = (float*)alloc(NYE * 4);                    // y / src2
  float* srcbuf = (float*)alloc(NYE * 4);                    // attn+resid / normalized
  bf16*  snbf   = (bf16*) alloc(NYE * 2);                    // normalized src, bf16
  char*  big    = (char*) alloc((size_t)MTOK * DFF * 2);     // ff (bf16) / xcat share
  bf16*  ffbf   = (bf16*)big;
  bf16*  xcat   = (bf16*)big;                                // lifetime-disjoint with ff
  bf16*  wembbf = (bf16*) alloc((size_t)(2 * DM) * KEMB * 2);
  bf16*  w1bf   = (bf16*) alloc((size_t)3 * DFF * DM * 2);
  bf16*  w2bf   = (bf16*) alloc((size_t)3 * DM * DFF * 2);
  float* part   = (float*)alloc((size_t)256 * 2 * DM * 4);
  float* acc    = (float*)alloc((size_t)2 * DM * 4);

  // ---- weight / input prep (bf16) ----
  prep_xcat_kernel<<<(unsigned)(((size_t)MEMB * KEMB) / 256), 256, 0, stream>>>(x_re, x_im, xcat);
  prep_wemb_kernel<<<(2 * DM * KEMB) / 256, 256, 0, stream>>>(eWre, eWim, wembbf);
  cvt_bf16_kernel<<<(3 * DFF * DM + 255) / 256, 256, 0, stream>>>(W1, w1bf, 3 * DFF * DM);
  cvt_bf16_kernel<<<(3 * DM * DFF + 255) / 256, 256, 0, stream>>>(W2, w2bf, 3 * DM * DFF);

  // ---- embedding GEMM: one N=512 WMMA GEMM for y_re and y_im ----
  gemm_emb_kernel<<<dim3(MEMB / 128, (2 * DM) / 128), 256, 0, stream>>>(
      xcat, wembbf, ebre, ebim, ybuf);

  const unsigned ew = (unsigned)(NYE / 256);  // exact: 110336 blocks
  for (int l = 0; l < 3; ++l) {
    attn_kernel<<<ew, 256, 0, stream>>>(ybuf, srcbuf);
    bn_reduce_kernel<<<256, 256, 0, stream>>>(srcbuf, part);
    bn_finalize_kernel<<<1, 256, 0, stream>>>(part, acc, 256);
    bn_apply_kernel<<<ew, 256, 0, stream>>>(srcbuf, acc, g1 + l * DM, be1 + l * DM, snbf);
    gemm_ff1_kernel<<<dim3(MTOK / 128, DFF / 128), 256, 0, stream>>>(
        snbf, w1bf + (size_t)l * DFF * DM, b1 + l * DFF, ffbf);
    gemm_ff2_kernel<<<dim3(MTOK / 128, DM / 128), 256, 0, stream>>>(
        ffbf, w2bf + (size_t)l * DM * DFF, b2 + l * DM, srcbuf, ybuf);
    bn_reduce_kernel<<<256, 256, 0, stream>>>(ybuf, part);
    bn_finalize_kernel<<<1, 256, 0, stream>>>(part, acc, 256);
    bn_apply_kernel<<<ew, 256, 0, stream>>>(ybuf, acc, g2 + l * DM, be2 + l * DM, (bf16*)nullptr);
  }

  write_out_kernel<<<(unsigned)(((size_t)BB * CC * DM) / 256), 256, 0, stream>>>(
      ybuf, (float*)d_out);
}
